// SelfAttention_9191230013677
// MI455X (gfx1250) — matrix operand order repaired and checked by reading, not compiled, not run on hardware
//
#include <hip/hip_runtime.h>
#include <hip/hip_bf16.h>

// ---------------------------------------------------------------------------
// Self-attention with summed-over-query output, CDNA5 (gfx1250) WMMA bf16.
//   result[b,:] = sum_k w[b,k] * V[b,k,:],  w[b,k] = sum_q softmax_k(E[b,q,:])
// Pipeline shape (all GEMMs): wave-private operand = 16 fragments hoisted in
// VGPRs (d<512; fits the 256 directly-addressable VGPRs, no spills) + the
// other d-half fetched from global (WGP$-resident) in odd stages; the
// block-shared operand streamed through double-buffered LDS via
// global_load_async_to_lds_b128 (ASYNCcnt), one barrier per stage.
// ---------------------------------------------------------------------------

#define DIMD    1024
#define SEQ     4096
#define BATCH   4
#define BS      (BATCH * SEQ)      // 16384 rows
#define HALF_D  512                // streamed d-chunk per stage
#define LSTR    520                // padded LDS row stride (elems); 1040B = 65*16B

typedef __attribute__((ext_vector_type(16))) __bf16          v16bf;
typedef __attribute__((ext_vector_type(16))) unsigned short  v16us;
typedef __attribute__((ext_vector_type(8)))  float           v8f;

// float -> bf16 (round to nearest even)
__device__ __forceinline__ unsigned short f2bf(float f) {
    unsigned int u = __float_as_uint(f);
    unsigned int r = u + 0x7FFFu + ((u >> 16) & 1u);
    return (unsigned short)(r >> 16);
}
__device__ __forceinline__ float bf2f(unsigned short h) {
    return __uint_as_float(((unsigned int)h) << 16);
}
__device__ __forceinline__ unsigned pack_bf16x2(float a, float b) {
#if __has_builtin(__builtin_amdgcn_cvt_pk_bf16_f32)
    auto v = __builtin_amdgcn_cvt_pk_bf16_f32(a, b);
    union { decltype(v) v2; unsigned u; } c; c.v2 = v;
    return c.u;
#else
    return (unsigned)f2bf(a) | ((unsigned)f2bf(b) << 16);
#endif
}

__device__ __forceinline__ v8f wmma_bf16(v16bf a, v16bf b, v8f c) {
    return __builtin_amdgcn_wmma_f32_16x16x32_bf16(
        false, a, false, b, (short)0, c, false, false);
}

// ---- CDNA5 async global->LDS copy (ASYNCcnt) ------------------------------
__device__ __forceinline__ unsigned lds_addr(void* p) {
    return (unsigned)(unsigned long long)(__attribute__((address_space(3))) void*)p;
}
__device__ __forceinline__ void async_ld_b128(unsigned ldsoff, const void* g) {
    asm volatile("global_load_async_to_lds_b128 %0, %1, off"
                 : : "v"(ldsoff), "v"((unsigned long long)g) : "memory");
}
__device__ __forceinline__ void wait_async_zero() {
    asm volatile("s_wait_asynccnt 0x0" : : : "memory");
}

// Cooperative async copy: 16 rows x 512 bf16 (16KB) into padded LDS tile.
// Chunk id = tid + 256*j -> row = (tid>>6) + 4*j, col = (tid&63)*8.
__device__ __forceinline__ void stage_issue(unsigned short (*tile)[LSTR],
                                            const unsigned short* g) {
    const int row = threadIdx.x >> 6;
    const int col = (threadIdx.x & 63) * 8;
    const unsigned short* gp = g + (size_t)row * DIMD + col;
    unsigned lo = lds_addr(&tile[row][col]);
#pragma unroll
    for (int j = 0; j < 4; ++j)
        async_ld_b128(lo + j * (4 * LSTR * 2), gp + (size_t)j * (4 * DIMD));
}

// ---- fragment loaders (contiguous k = 16*(lane/16) + s per lane, wave32) ------------------------
// A fragment from global, base precomputed: pbase = M + (row0 + lane%16)*ld + 16*(lane/16)
__device__ __forceinline__ v16bf frag_a_g(const unsigned short* pbase, int d0) {
    union { v16us us; v16bf bf; uint4 q[2]; } u;
    u.q[0] = *(const uint4*)(pbase + d0);
    u.q[1] = *(const uint4*)(pbase + d0 + 8);
    return u.bf;
}
// B fragment from global, base precomputed: pbase = M + (col0 + lane%16)*ld + 16*(lane/16)
__device__ __forceinline__ v16bf frag_b_g(const unsigned short* pbase, int d0) {
    union { v16us us; v16bf bf; uint4 q[2]; } u;
    u.q[0] = *(const uint4*)(pbase + d0);
    u.q[1] = *(const uint4*)(pbase + d0 + 8);
    return u.bf;
}
// B fragment from a 16-row padded LDS tile (rows = B columns).
__device__ __forceinline__ v16bf load_b_lds(const unsigned short* tile,
                                            int d0, int lane) {
    int l = lane & 15, h = lane >> 4;
    const unsigned short* p = tile + l * LSTR + d0 + 16 * h;
    union { v16us us; v16bf bf; uint4 q[2]; } u;
    u.q[0] = ((const uint4*)p)[0];
    u.q[1] = ((const uint4*)p)[1];
    return u.bf;
}
// A fragment from a 16-row padded LDS tile (rows = A rows).
__device__ __forceinline__ v16bf load_a_lds(const unsigned short* tile,
                                            int d0, int lane) {
    int l = lane & 15, h = lane >> 4;
    const unsigned short* p = tile + l * LSTR + d0 + 16 * h;
    union { v16us us; v16bf bf; uint4 q[2]; } u;
    u.q[0] = *(const uint4*)p;
    u.q[1] = *(const uint4*)(p + 8);
    return u.bf;
}

__device__ __forceinline__ float redmax16(float v) {
#pragma unroll
    for (int m = 1; m < 16; m <<= 1) v = fmaxf(v, __shfl_xor(v, m, 32));
    return v;
}
__device__ __forceinline__ float redsum16(float v) {
#pragma unroll
    for (int m = 1; m < 16; m <<= 1) v += __shfl_xor(v, m, 32);
    return v;
}

// ---------------------------------------------------------------------------
// Prep 1: X fp32 -> bf16 (row-major copy-convert).
// ---------------------------------------------------------------------------
__global__ void __launch_bounds__(256)
cvt_x_kernel(const float* __restrict__ x, unsigned short* __restrict__ o) {
    size_t i4 = ((size_t)blockIdx.x * 256 + threadIdx.x) * 4;
    float4 f = *(const float4*)(x + i4);
    uint2 p;
    p.x = pack_bf16x2(f.x, f.y);
    p.y = pack_bf16x2(f.z, f.w);
    *(uint2*)(o + i4) = p;
}

// ---------------------------------------------------------------------------
// Prep 2: W fp32 [d][n] -> WT bf16 [n][d] (LDS 32x32 tile transpose).
// ---------------------------------------------------------------------------
__global__ void __launch_bounds__(256)
wt_kernel(const float* __restrict__ W, unsigned short* __restrict__ WT) {
    __shared__ float t[32][33];
    const int bn = blockIdx.x * 32;
    const int bd = blockIdx.y * 32;
    const int c  = threadIdx.x & 31;
    const int r0 = threadIdx.x >> 5;
#pragma unroll
    for (int i = 0; i < 4; ++i) {
        int r = r0 + 8 * i;
        t[r][c] = W[(size_t)(bd + r) * DIMD + bn + c];
    }
    __syncthreads();
#pragma unroll
    for (int i = 0; i < 4; ++i) {
        int r = r0 + 8 * i;
        WT[(size_t)(bn + r) * DIMD + bd + c] = f2bf(t[c][r]);
    }
}

// ---------------------------------------------------------------------------
// Kernel 1: C = (Xb @ WT^T + bias) * scale, bf16 out (all operands bf16).
// ---------------------------------------------------------------------------
__global__ void __launch_bounds__(256, 1)
qkv_proj_kernel(const unsigned short* __restrict__ Xb,
                const unsigned short* __restrict__ WTb,
                const float* __restrict__ bias, unsigned short* __restrict__ outb,
                float scale) {
    __shared__ __align__(16) unsigned short Xs[2][16][LSTR];

    const int wave = threadIdx.x >> 5;
    const int lane = threadIdx.x & 31;
    const int l = lane & 15, h = lane >> 4;
    const int n0    = blockIdx.x * 128 + wave * 16;
    const int mbase = blockIdx.y * 512;            // 32 m-tiles per block

    const unsigned short* bBase = WTb + (size_t)(n0 + l) * DIMD + 16 * h;
    v16bf bw[16];
#pragma unroll
    for (int i = 0; i < 16; ++i) bw[i] = frag_b_g(bBase, i * 32);
    const float bv = bias[n0 + l];

    const v8f zero = {};
    v8f acc = zero;
    const int NST = 32 * 2;

    stage_issue(Xs[0], Xb + (size_t)mbase * DIMD);
    for (int s = 0; s < NST; ++s) {
        wait_async_zero();
        __syncthreads();
        if (s + 1 < NST) {
            int mt = (s + 1) >> 1, dh = (s + 1) & 1;
            stage_issue(Xs[(s + 1) & 1],
                        Xb + (size_t)(mbase + mt * 16) * DIMD + dh * HALF_D);
        }
        const unsigned short* tile = &Xs[s & 1][0][0];
        if ((s & 1) == 0) {
#pragma unroll
            for (int dc = 0; dc < 16; ++dc)
                acc = wmma_bf16(load_a_lds(tile, dc * 32, lane), bw[dc], acc);
        } else {
#pragma unroll
            for (int dc = 0; dc < 16; ++dc)
                acc = wmma_bf16(load_a_lds(tile, dc * 32, lane),
                                frag_b_g(bBase, HALF_D + dc * 32), acc);
            int m0 = mbase + (s >> 1) * 16;
#pragma unroll
            for (int r = 0; r < 8; ++r) {
                float v = (acc[r] + bv) * scale;
                outb[(size_t)(m0 + r + 8 * h) * DIMD + n0 + l] = f2bf(v);
            }
            acc = zero;
        }
    }
}

// ---------------------------------------------------------------------------
// Kernel 2: softmax row stats of E = Q K^T (k range split in halves).
// ---------------------------------------------------------------------------
__global__ void __launch_bounds__(256, 1)
attn_stats_kernel(const unsigned short* __restrict__ Qb,
                  const unsigned short* __restrict__ Kb,
                  float* __restrict__ mpart, float* __restrict__ lpart) {
    __shared__ __align__(16) unsigned short Ks[2][16][LSTR];

    const int wave = threadIdx.x >> 5;
    const int lane = threadIdx.x & 31;
    const int h = lane >> 4;

    const int qt    = (blockIdx.x >> 1) * 8 + wave;
    const int half  = blockIdx.x & 1;
    const int q0    = qt * 16;
    const int kbase = (q0 >> 12) << 12;     // batch * SEQ
    const int kt0   = half * (SEQ / 32);    // 128 k-tiles per half

    const unsigned short* aBase = Qb + (size_t)(q0 + (lane & 15)) * DIMD + 16 * h;
    v16bf aq[16];
#pragma unroll
    for (int i = 0; i < 16; ++i) aq[i] = frag_a_g(aBase, i * 32);

    float m_[8], l_[8];
#pragma unroll
    for (int r = 0; r < 8; ++r) { m_[r] = -3.0e38f; l_[r] = 0.0f; }

    const v8f zero = {};
    v8f acc = zero;
    const int NST = (SEQ / 32) * 2;

    stage_issue(Ks[0], Kb + (size_t)(kbase + kt0 * 16) * DIMD);
    for (int s = 0; s < NST; ++s) {
        wait_async_zero();
        __syncthreads();
        if (s + 1 < NST) {
            int kt = kt0 + ((s + 1) >> 1);
            int dh = (s + 1) & 1;
            stage_issue(Ks[(s + 1) & 1],
                        Kb + (size_t)(kbase + kt * 16) * DIMD + dh * HALF_D);
        }
        const unsigned short* tile = &Ks[s & 1][0][0];
        if ((s & 1) == 0) {
#pragma unroll
            for (int dc = 0; dc < 16; ++dc)
                acc = wmma_bf16(aq[dc], load_b_lds(tile, dc * 32, lane), acc);
        } else {
#pragma unroll
            for (int dc = 0; dc < 16; ++dc)
                acc = wmma_bf16(frag_a_g(aBase, HALF_D + dc * 32),
                                load_b_lds(tile, dc * 32, lane), acc);
#pragma unroll
            for (int r = 0; r < 8; ++r) {
                float e  = acc[r];
                float nm = fmaxf(m_[r], redmax16(e));
                float p  = __expf(e - nm);
                float rs = redsum16(p);
                l_[r] = l_[r] * __expf(m_[r] - nm) + rs;
                m_[r] = nm;
            }
            acc = zero;
        }
    }

    if ((lane & 15) == 0) {
#pragma unroll
        for (int r = 0; r < 8; ++r) {
            int q = q0 + r + 8 * h;
            mpart[half * BS + q] = m_[r];
            lpart[half * BS + q] = l_[r];
        }
    }
}

// Merge the two k-half stats (deterministic).
__global__ void __launch_bounds__(256)
stats_combine_kernel(const float* __restrict__ mpart, const float* __restrict__ lpart,
                     float* __restrict__ mbuf, float* __restrict__ lbuf) {
    int i = blockIdx.x * 256 + threadIdx.x;
    float m0 = mpart[i], m1 = mpart[BS + i];
    float l0 = lpart[i], l1 = lpart[BS + i];
    float m  = fmaxf(m0, m1);
    lbuf[i]  = l0 * __expf(m0 - m) + l1 * __expf(m1 - m);
    mbuf[i]  = m;
}

// ---------------------------------------------------------------------------
// Kernel 3: attention column sums w[k] = sum_q exp(E-m)/l (q range split).
// ---------------------------------------------------------------------------
__global__ void __launch_bounds__(256, 1)
attn_colsum_kernel(const unsigned short* __restrict__ Qb,
                   const unsigned short* __restrict__ Kb,
                   const float* __restrict__ mbuf, const float* __restrict__ lbuf,
                   float* __restrict__ wpart) {
    __shared__ __align__(16) unsigned short Qs[2][16][LSTR];

    const int wave = threadIdx.x >> 5;
    const int lane = threadIdx.x & 31;
    const int h = lane >> 4;

    const int kt    = (blockIdx.x >> 1) * 8 + wave;
    const int half  = blockIdx.x & 1;
    const int k0    = kt * 16;
    const int qbase = (k0 >> 12) << 12;
    const int qt0   = half * (SEQ / 32);

    const unsigned short* bBase = Kb + (size_t)(k0 + (lane & 15)) * DIMD + 16 * h;
    v16bf bk[16];
#pragma unroll
    for (int i = 0; i < 16; ++i) bk[i] = frag_b_g(bBase, i * 32);

    const v8f zero = {};
    v8f acc = zero;
    float wacc = 0.0f;
    const int NST = (SEQ / 32) * 2;

    stage_issue(Qs[0], Qb + (size_t)(qbase + qt0 * 16) * DIMD);
    for (int s = 0; s < NST; ++s) {
        wait_async_zero();
        __syncthreads();
        if (s + 1 < NST) {
            int qt = qt0 + ((s + 1) >> 1);
            int dh = (s + 1) & 1;
            stage_issue(Qs[(s + 1) & 1],
                        Qb + (size_t)(qbase + qt * 16) * DIMD + dh * HALF_D);
        }
        const unsigned short* tile = &Qs[s & 1][0][0];
        if ((s & 1) == 0) {
#pragma unroll
            for (int dc = 0; dc < 16; ++dc)
                acc = wmma_bf16(load_a_lds(tile, dc * 32, lane), bk[dc], acc);
        } else {
#pragma unroll
            for (int dc = 0; dc < 16; ++dc)
                acc = wmma_bf16(load_a_lds(tile, dc * 32, lane),
                                frag_b_g(bBase, HALF_D + dc * 32), acc);
            int q0 = qbase + (qt0 + (s >> 1)) * 16;
#pragma unroll
            for (int r = 0; r < 8; ++r) {
                int q = q0 + r + 8 * h;
                wacc += __expf(acc[r] - mbuf[q]) / lbuf[q];
            }
            acc = zero;
        }
    }

    wacc += __shfl_xor(wacc, 16, 32);
    if (lane < 16) wpart[half * BS + k0 + lane] = wacc;
}

// ---------------------------------------------------------------------------
// Kernel 4: out[b,d] = sum_k (w0[b,k]+w1[b,k]) * V[b,k,d]
// ---------------------------------------------------------------------------
__global__ void __launch_bounds__(256)
attn_out_kernel(const unsigned short* __restrict__ Vb,
                const float* __restrict__ wpart, float* __restrict__ out) {
    const int b = blockIdx.y;
    const int d = blockIdx.x * 256 + threadIdx.x;
    const unsigned short* vp = Vb + ((size_t)b * SEQ) * DIMD + d;
    const float* wp0 = wpart + b * SEQ;
    const float* wp1 = wpart + BS + b * SEQ;

    float acc = 0.0f;
#pragma unroll 4
    for (int k = 0; k < SEQ; ++k) {
        acc += (wp0[k] + wp1[k]) * bf2f(vp[(size_t)k * DIMD]);
    }
    out[b * DIMD + d] = acc;
}

// ---------------------------------------------------------------------------
extern "C" void kernel_launch(void* const* d_in, const int* in_sizes, int n_in,
                              void* d_out, int out_size, void* d_ws, size_t ws_size,
                              hipStream_t stream) {
    (void)in_sizes; (void)n_in; (void)out_size; (void)ws_size;

    const float* x  = (const float*)d_in[0];
    const float* Wq = (const float*)d_in[1];
    const float* bq = (const float*)d_in[2];
    const float* Wk = (const float*)d_in[3];
    const float* bk = (const float*)d_in[4];
    const float* Wv = (const float*)d_in[5];
    const float* bv = (const float*)d_in[6];
    float* out = (float*)d_out;

    unsigned short* Qb  = (unsigned short*)d_ws;
    unsigned short* Kb  = Qb  + (size_t)BS * DIMD;
    unsigned short* Vb  = Kb  + (size_t)BS * DIMD;
    unsigned short* Xb  = Vb  + (size_t)BS * DIMD;
    unsigned short* WTq = Xb  + (size_t)BS * DIMD;
    unsigned short* WTk = WTq + (size_t)DIMD * DIMD;
    unsigned short* WTv = WTk + (size_t)DIMD * DIMD;
    float* mbuf  = (float*)(WTv + (size_t)DIMD * DIMD);
    float* lbuf  = mbuf + BS;
    float* mpart = lbuf + BS;          // [2][BS]
    float* lpart = mpart + 2 * BS;     // [2][BS]
    float* wpart = lpart + 2 * BS;     // [2][BS]

    const float scale = 0.03125f;      // D^-0.5 folded into Q

    cvt_x_kernel<<<(size_t)BS * DIMD / 1024, 256, 0, stream>>>(x, Xb);
    dim3 gWT(DIMD / 32, DIMD / 32);
    wt_kernel<<<gWT, 256, 0, stream>>>(Wq, WTq);
    wt_kernel<<<gWT, 256, 0, stream>>>(Wk, WTk);
    wt_kernel<<<gWT, 256, 0, stream>>>(Wv, WTv);

    dim3 gProj(DIMD / 128, BS / 512);
    qkv_proj_kernel<<<gProj, 256, 0, stream>>>(Xb, WTq, bq, Qb, scale);
    qkv_proj_kernel<<<gProj, 256, 0, stream>>>(Xb, WTk, bk, Kb, 1.0f);
    qkv_proj_kernel<<<gProj, 256, 0, stream>>>(Xb, WTv, bv, Vb, 1.0f);

    attn_stats_kernel   <<<BS / 64, 256, 0, stream>>>(Qb, Kb, mpart, lpart);
    stats_combine_kernel<<<BS / 256, 256, 0, stream>>>(mpart, lpart, mbuf, lbuf);
    attn_colsum_kernel  <<<BS / 64, 256, 0, stream>>>(Qb, Kb, mbuf, lbuf, wpart);

    dim3 gOut(DIMD / 256, BATCH);
    attn_out_kernel<<<gOut, 256, 0, stream>>>(Vb, wpart, out);
}
